// Attention_36326833390265
// MI455X (gfx1250) — matrix operand order repaired and checked by reading, compiled, not run on hardware
//
#include <hip/hip_runtime.h>
#include <stdint.h>

#define BB 8
#define CC 256
#define NN 16384   // 128*128

typedef __attribute__((ext_vector_type(16))) __bf16 v16bf;
typedef __attribute__((ext_vector_type(8)))  float  v8f;

struct U4x2 { uint4 lo; uint4 hi; };

// Build a 16xbf16 WMMA fragment from two 16-byte chunks (LDS or global).
__device__ inline v16bf frag_from(const void* p_lo, const void* p_hi) {
    U4x2 u;
    u.lo = *(const uint4*)p_lo;
    u.hi = *(const uint4*)p_hi;
    return __builtin_bit_cast(v16bf, u);
}

__device__ inline unsigned short bfbits(float f) {
    __bf16 h = (__bf16)f;
    return __builtin_bit_cast(unsigned short, h);
}
__device__ inline uint32_t pk2(float a, float b) {
    return (uint32_t)bfbits(a) | ((uint32_t)bfbits(b) << 16);
}

// ---------------------------------------------------------------- zero
__global__ void k_zero(float* __restrict__ p, int n) {
    int i = blockIdx.x * blockDim.x + threadIdx.x;
    if (i < n) p[i] = 0.f;
}

// ------------------------------------------------------- row sums s = X*1
__global__ __launch_bounds__(256) void k_rowsum(const float* __restrict__ x,
                                                float* __restrict__ s) {
    const int row = blockIdx.x;                 // b*C + c
    const float4* xr = (const float4*)(x + (size_t)row * NN);
    float acc = 0.f;
    for (int i = threadIdx.x; i < NN / 4; i += 256) {
        float4 v = xr[i];
        acc += v.x + v.y + v.z + v.w;
    }
    __shared__ float red[256];
    red[threadIdx.x] = acc;
    __syncthreads();
    for (int off = 128; off > 0; off >>= 1) {
        if (threadIdx.x < off) red[threadIdx.x] += red[threadIdx.x + off];
        __syncthreads();
    }
    if (threadIdx.x == 0) s[row] = red[0];
}

// -------------------------------------------- Gram: G += X[128 rows] * X^T
// grid: (4 tiles of 128x128, B, 16 k-splits of 1024). 256 threads = 8 waves.
// Wave w owns subtile row w (8 subtiles, shared A fragment), double-buffered LDS.
__global__ __launch_bounds__(256) void k_gram(const float* __restrict__ x,
                                              float* __restrict__ G) {
    const int ti = blockIdx.x;            // 0..3
    const int b  = blockIdx.y;
    const int kz = blockIdx.z;            // 0..15
    const int tile_r = (ti >> 1) * 128;
    const int tile_c = (ti & 1) * 128;
    const int kbase  = kz * 1024;

    __shared__ __bf16 As[2][128 * 32];
    __shared__ __bf16 Bs[2][128 * 32];

    const int t     = threadIdx.x;
    const int lane  = t & 31;
    const int wave  = t >> 5;
    const int lhalf = lane >> 4;          // 0: lanes 0-15, 1: lanes 16-31
    const int lmod  = lane & 15;

    // Fragment element offsets per ISA 16-bit layouts
    const int a_row = wave * 16 + lmod;
    const int a_k0  = lhalf ? 8 : 0;      // A: K {0..7,16..23} / {8..15,24..31}
    const int b_k0  = lhalf ? 8 : 0;      // B: K {0..7,16..23} / {8..15,24..31}

    v8f acc[8];
    #pragma unroll
    for (int i = 0; i < 8; ++i) acc[i] = (v8f){0.f,0.f,0.f,0.f,0.f,0.f,0.f,0.f};

    const float* xb = x + (size_t)b * CC * NN;

    // Per-thread staging slots: 1024 float4-slots per operand, 4 per thread.
    int srow[4], sk4[4];
    #pragma unroll
    for (int s2 = 0; s2 < 4; ++s2) {
        int slot  = t + s2 * 256;
        srow[s2] = slot >> 3;             // 0..127
        sk4[s2]  = (slot & 7) * 4;        // 0..28
    }

    // Prologue: stage k-step 0 into buffer 0.
    #pragma unroll
    for (int s2 = 0; s2 < 4; ++s2) {
        float4 av = *(const float4*)&xb[(size_t)(tile_r + srow[s2]) * NN + kbase + sk4[s2]];
        float4 bv = *(const float4*)&xb[(size_t)(tile_c + srow[s2]) * NN + kbase + sk4[s2]];
        *(uint2*)&As[0][srow[s2] * 32 + sk4[s2]] = make_uint2(pk2(av.x, av.y), pk2(av.z, av.w));
        *(uint2*)&Bs[0][srow[s2] * 32 + sk4[s2]] = make_uint2(pk2(bv.x, bv.y), pk2(bv.z, bv.w));
    }

    int cur = 0;
    for (int it = 0; it < 32; ++it) {
        __syncthreads();                  // single barrier per k-step
        const int k0 = kbase + it * 32;
        const bool more = (it + 1 < 32);

        // Issue next tile's global loads before the WMMAs (latency overlap),
        // and prefetch two k-steps ahead into the cache hierarchy.
        float4 avn[4], bvn[4];
        if (more) {
            #pragma unroll
            for (int s2 = 0; s2 < 4; ++s2) {
                const float* pa = &xb[(size_t)(tile_r + srow[s2]) * NN + k0 + 32 + sk4[s2]];
                const float* pb = &xb[(size_t)(tile_c + srow[s2]) * NN + k0 + 32 + sk4[s2]];
                avn[s2] = *(const float4*)pa;
                bvn[s2] = *(const float4*)pb;
                if (it + 2 < 32) {
                    __builtin_prefetch(pa + 32, 0, 3);
                    __builtin_prefetch(pb + 32, 0, 3);
                }
            }
        }

        // Compute on current buffer: 1 A-fragment reused across 8 B-fragments.
        const __bf16* Ac = As[cur];
        const __bf16* Bc = Bs[cur];
        v16bf af = frag_from(&Ac[a_row * 32 + a_k0], &Ac[a_row * 32 + a_k0 + 16]);
        #pragma unroll
        for (int j = 0; j < 8; ++j) {
            const int br = (j * 16 + lmod) * 32;
            v16bf bfj = frag_from(&Bc[br + b_k0], &Bc[br + b_k0 + 16]);
            acc[j] = __builtin_amdgcn_wmma_f32_16x16x32_bf16(false, af, false, bfj,
                                                             (short)0, acc[j], false, false);
        }

        // Pack next tile into the other buffer (no extra barrier needed).
        if (more) {
            __bf16* An = As[cur ^ 1];
            __bf16* Bn = Bs[cur ^ 1];
            #pragma unroll
            for (int s2 = 0; s2 < 4; ++s2) {
                *(uint2*)&An[srow[s2] * 32 + sk4[s2]] =
                    make_uint2(pk2(avn[s2].x, avn[s2].y), pk2(avn[s2].z, avn[s2].w));
                *(uint2*)&Bn[srow[s2] * 32 + sk4[s2]] =
                    make_uint2(pk2(bvn[s2].x, bvn[s2].y), pk2(bvn[s2].z, bvn[s2].w));
            }
        }
        cur ^= 1;
    }

    // Split-K accumulation into G.
    float* Gb = G + (size_t)b * CC * CC;
    const int gm0 = tile_r + wave * 16 + lhalf * 8;
    #pragma unroll
    for (int j = 0; j < 8; ++j) {
        const int gn = tile_c + j * 16 + lmod;
        #pragma unroll
        for (int r = 0; r < 8; ++r) {
            atomicAdd(&Gb[(size_t)(gm0 + r) * CC + gn], acc[j][r]);
        }
    }
}

// ---------------------------------------------- tiny GEMM: T = G * w2^T
__global__ __launch_bounds__(256) void k_gw2(const float* __restrict__ G,
                                             const float* __restrict__ w2,
                                             float* __restrict__ T) {
    const int bc = blockIdx.x;            // b*C + c
    const int d  = threadIdx.x;
    __shared__ float g[256];
    g[d] = G[(size_t)bc * CC + d];
    __syncthreads();
    float acc = 0.f;
    const float* w2d = w2 + (size_t)d * CC;
    for (int e = 0; e < CC; ++e) acc += g[e] * w2d[e];
    T[(size_t)bc * CC + d] = acc;
}

// ---------- logits = w1*T + (w1 s) b2^T + b1 (w2 s)^T + N b1 b2^T ; softmax
__global__ __launch_bounds__(256) void k_logits_softmax(
    const float* __restrict__ T,  const float* __restrict__ w1,
    const float* __restrict__ b1, const float* __restrict__ b2,
    const float* __restrict__ w2, const float* __restrict__ s,
    unsigned short* __restrict__ A) {
    const int bo = blockIdx.x;            // b*C + o
    const int b  = bo >> 8, o = bo & 255;
    const int d  = threadIdx.x;
    __shared__ float red[256];
    const float* sb = s + b * CC;

    // w1s = dot(w1[o], s_b)  (block reduction)
    red[d] = w1[(size_t)o * CC + d] * sb[d];
    __syncthreads();
    for (int off = 128; off > 0; off >>= 1) {
        if (d < off) red[d] += red[d + off];
        __syncthreads();
    }
    float w1s = red[0];
    __syncthreads();

    // w2s[d] = dot(w2[d], s_b)  (per-thread)
    float w2s = 0.f;
    const float* w2d = w2 + (size_t)d * CC;
    for (int e = 0; e < CC; ++e) w2s += w2d[e] * sb[e];

    // main: sum_c w1[o][c] * T[b][c][d]
    float v = 0.f;
    const float* Tb  = T + (size_t)b * CC * CC;
    const float* w1o = w1 + (size_t)o * CC;
    for (int c = 0; c < CC; ++c) v += w1o[c] * Tb[(size_t)c * CC + d];
    v += w1s * b2[d] + b1[o] * w2s + (float)NN * b1[o] * b2[d];

    // softmax over d
    red[d] = v; __syncthreads();
    for (int off = 128; off > 0; off >>= 1) {
        if (d < off) red[d] = fmaxf(red[d], red[d + off]);
        __syncthreads();
    }
    float mx = red[0]; __syncthreads();
    float e  = __expf(v - mx);
    red[d] = e; __syncthreads();
    for (int off = 128; off > 0; off >>= 1) {
        if (d < off) red[d] += red[d + off];
        __syncthreads();
    }
    A[(size_t)bo * CC + d] = bfbits(e / red[0]);
}

// ------------------------------------ out = x + A(bf16) * X, WMMA over K=256
// grid: (N/64 column chunks, B). 8 waves: wave = N-subtile (w&3), M-half (w>>2).
// X staged double-buffered into LDS; A fragments straight from L2.
__global__ __launch_bounds__(256) void k_av(const float* __restrict__ x,
                                            const unsigned short* __restrict__ Araw,
                                            float* __restrict__ out) {
    const int cb   = blockIdx.x;
    const int b    = blockIdx.y;
    const int col0 = cb * 64;

    __shared__ __bf16 Bs[2][64 * 32];     // Bs[buf][col*32 + k], fragment-ready

    const int t     = threadIdx.x;
    const int lane  = t & 31;
    const int wave  = t >> 5;
    const int ns    = wave & 3;           // N subtile 0..3
    const int mh    = wave >> 2;          // M half: subtiles mh*8 .. mh*8+7
    const int lhalf = lane >> 4;
    const int lmod  = lane & 15;

    v8f acc[8];
    #pragma unroll
    for (int i = 0; i < 8; ++i) acc[i] = (v8f){0.f,0.f,0.f,0.f,0.f,0.f,0.f,0.f};

    const float* xb = x + (size_t)b * CC * NN;
    const __bf16* Ab = (const __bf16*)Araw + (size_t)b * CC * CC;

    const int a_k0 = lhalf ? 8 : 0;
    const int b_k0 = lhalf ? 8 : 0;

    const int dA = t >> 4;                // 0..15
    const int c4 = (t & 15) * 4;          // 0..60

    // Prologue: stage d-step 0 into buffer 0.
    #pragma unroll
    for (int sl = 0; sl < 2; ++sl) {
        int d = dA + sl * 16;
        float4 v = *(const float4*)&xb[(size_t)d * NN + col0 + c4];
        Bs[0][(c4 + 0) * 32 + d] = (__bf16)v.x;
        Bs[0][(c4 + 1) * 32 + d] = (__bf16)v.y;
        Bs[0][(c4 + 2) * 32 + d] = (__bf16)v.z;
        Bs[0][(c4 + 3) * 32 + d] = (__bf16)v.w;
    }

    int cur = 0;
    for (int d0 = 0; d0 < CC; d0 += 32) {
        __syncthreads();
        const bool more = (d0 + 32 < CC);

        float4 nv[2];
        if (more) {
            #pragma unroll
            for (int sl = 0; sl < 2; ++sl) {
                int d = d0 + 32 + dA + sl * 16;
                nv[sl] = *(const float4*)&xb[(size_t)d * NN + col0 + c4];
            }
        }

        const int brow = (ns * 16 + lmod) * 32;
        v16bf bf = frag_from(&Bs[cur][brow + b_k0], &Bs[cur][brow + b_k0 + 16]);
        #pragma unroll
        for (int i = 0; i < 8; ++i) {
            int row = (mh * 8 + i) * 16 + lmod;
            v16bf af = frag_from(&Ab[(size_t)row * CC + d0 + a_k0],
                                 &Ab[(size_t)row * CC + d0 + a_k0 + 16]);
            acc[i] = __builtin_amdgcn_wmma_f32_16x16x32_bf16(false, af, false, bf,
                                                             (short)0, acc[i], false, false);
        }

        if (more) {
            __bf16* Bn = Bs[cur ^ 1];
            #pragma unroll
            for (int sl = 0; sl < 2; ++sl) {
                int d = dA + sl * 16;
                Bn[(c4 + 0) * 32 + d] = (__bf16)nv[sl].x;
                Bn[(c4 + 1) * 32 + d] = (__bf16)nv[sl].y;
                Bn[(c4 + 2) * 32 + d] = (__bf16)nv[sl].z;
                Bn[(c4 + 3) * 32 + d] = (__bf16)nv[sl].w;
            }
        }
        cur ^= 1;
    }

    float* ob = out + (size_t)b * CC * NN;
    const int n = col0 + ns * 16 + lmod;
    #pragma unroll
    for (int i = 0; i < 8; ++i) {
        int m0 = (mh * 8 + i) * 16 + lhalf * 8;
        #pragma unroll
        for (int r = 0; r < 8; ++r) {
            size_t idx = (size_t)(m0 + r) * NN + n;
            ob[idx] = xb[idx] + acc[i][r];
        }
    }
}

// ----------------------------------------------------------------- launch
extern "C" void kernel_launch(void* const* d_in, const int* in_sizes, int n_in,
                              void* d_out, int out_size, void* d_ws, size_t ws_size,
                              hipStream_t stream) {
    (void)in_sizes; (void)n_in; (void)out_size; (void)ws_size;
    const float* x  = (const float*)d_in[0];
    const float* w1 = (const float*)d_in[1];
    const float* b1 = (const float*)d_in[2];
    const float* w2 = (const float*)d_in[3];
    const float* b2 = (const float*)d_in[4];
    float* out = (float*)d_out;

    // Workspace layout (~5.3 MB): G, T (f32 B*C*C each), s (B*C), A (bf16 B*C*C)
    float* G = (float*)d_ws;
    float* T = G + (size_t)BB * CC * CC;
    float* s = T + (size_t)BB * CC * CC;
    unsigned short* A = (unsigned short*)(s + BB * CC);

    int ng = BB * CC * CC;
    k_zero<<<(ng + 255) / 256, 256, 0, stream>>>(G, ng);
    k_rowsum<<<BB * CC, 256, 0, stream>>>(x, s);
    dim3 gg(4, BB, 16);
    k_gram<<<gg, 256, 0, stream>>>(x, G);
    k_gw2<<<BB * CC, 256, 0, stream>>>(G, w2, T);
    k_logits_softmax<<<BB * CC, 256, 0, stream>>>(T, w1, b1, b2, w2, s, A);
    dim3 g3(NN / 64, BB, 1);
    k_av<<<g3, 256, 0, stream>>>(x, A, out);
}
